// DeformableTransformerEncoderLayer_17085379903978
// MI455X (gfx1250) — matrix operand order repaired and checked by reading, not compiled, not run on hardware
//
#include <hip/hip_runtime.h>
#include <hip/hip_bf16.h>
#include <stdint.h>

// ---------------- problem constants (from reference) ----------------
#define B_    4
#define LQ_   13343
#define M_    (B_*LQ_)        // 53372 tokens
#define MP_   53376           // M padded to multiple of 32
#define D_    256
#define H_    8
#define HD_   32
#define L_    5
#define P_    4
#define DFF_  2048
#define NP_OFF 320            // H*L*P*2, already /64
#define NF_ATT 160            // H*L*P
#define NP_ATT 192            // padded to /64 (stride of attn buffer)

typedef _Float16 v16h __attribute__((ext_vector_type(16)));
typedef _Float16 v8h  __attribute__((ext_vector_type(8)));
typedef float    v8f  __attribute__((ext_vector_type(8)));

// A fragment (16-bit A 16x32): per lane, halves at k {0..7} and {16..23} (+8 if lane>=16)
__device__ __forceinline__ v16h load_afrag(const _Float16* __restrict__ p)
{
    v8h lo = *(const v8h*)p;
    v8h hi = *(const v8h*)(p + 16);
    return __builtin_shufflevector(lo, hi, 0,1,2,3,4,5,6,7,8,9,10,11,12,13,14,15);
}
// B fragment (16-bit B 32x16): per lane, halves at k {0..7} and {16..23} (+8 if lane>=16)
__device__ __forceinline__ v16h load_bfrag(const _Float16* __restrict__ p)
{
    v8h lo = *(const v8h*)p;
    v8h hi = *(const v8h*)(p + 16);
    return __builtin_shufflevector(lo, hi, 0,1,2,3,4,5,6,7,8,9,10,11,12,13,14,15);
}

#define WMMA_F16(a, b, c) \
    __builtin_amdgcn_wmma_f32_16x16x32_f16(false, (a), false, (b), (short)0, (c), false, false)

// ---------------- weight convert + transpose (fp32 [K,Nfull] -> f16 [NP,K]) --
__global__ void convw_kernel(const float* __restrict__ W, _Float16* __restrict__ Wt,
                             int K, int Nfull, int NP)
{
    int idx = blockIdx.x * blockDim.x + threadIdx.x;
    if (idx >= NP * K) return;
    int n = idx / K;
    int k = idx - n * K;
    Wt[idx] = (n < Nfull) ? (_Float16)W[(size_t)k * Nfull + n] : (_Float16)0.f;
}

// ---------------- prep: q16 = f16(src+pos), s16 = f16(src), zero pad rows ----
__global__ void prep_kernel(const float* __restrict__ src, const float* __restrict__ pos,
                            _Float16* __restrict__ q16, _Float16* __restrict__ s16)
{
    size_t i = (size_t)blockIdx.x * blockDim.x + threadIdx.x;
    if (i >= (size_t)MP_ * D_) return;
    if (i < (size_t)M_ * D_) {
        float s = src[i];
        q16[i] = (_Float16)(s + pos[i]);
        s16[i] = (_Float16)s;
    } else {
        q16[i] = (_Float16)0.f;
        s16[i] = (_Float16)0.f;
    }
}

// ---------------- WMMA GEMM: C[MP,NP] = A[MP,K] @ Bt[NP,K]^T + bias ---------
// Wave tile 32(M) x 64(N): 2 A fragments, 4 B fragments (shared by both M
// halves), 8 accumulators. K and NP are compile-time so all six fragment
// addresses are immediate offsets from two loop-carried base pointers.
template <int K, int NP, bool RELU, bool OUTF16>
__global__ __attribute__((amdgpu_waves_per_eu(1)))
void gemm_wmma_kernel(const _Float16* __restrict__ A,
                      const _Float16* __restrict__ Bt,
                      const float* __restrict__ bias,
                      float* __restrict__ Cf,
                      _Float16* __restrict__ Ch,
                      int Nfull, int totalWaves)
{
    const int wv = (blockIdx.x * blockDim.x + threadIdx.x) >> 5;   // wave-uniform
    if (wv >= totalWaves) return;                                   // uniform exit
    const int lane = threadIdx.x & 31;
    constexpr int NT = NP >> 6;
    const int mt = wv / NT;
    const int nt = wv - mt * NT;

    const int mrow  = lane & 15;
    const int ahalf = (lane & 16) ? 8  : 0;       // A K sub-offset
    const int bhalf = (lane & 16) ? 8  : 0;       // B K sub-offset
    const int col0  = (nt << 6) + mrow;

    const _Float16* __restrict__ ap = A  + (size_t)((mt << 5) + mrow) * K + ahalf;
    const _Float16* __restrict__ bq = Bt + (size_t)((nt << 6) + mrow) * K + bhalf;

    v8f acc[2][4];
#pragma unroll
    for (int s = 0; s < 4; ++s) {
        int c = col0 + s * 16;
        float bv = (c < Nfull) ? bias[c] : 0.f;
#pragma unroll
        for (int r = 0; r < 8; ++r) { acc[0][s][r] = bv; acc[1][s][r] = bv; }
    }

#pragma unroll 2
    for (int k0 = 0; k0 < K; k0 += 32) {
        // all fragment loads issue as one clause; compiler inserts partial waits
        v16h a0 = load_afrag(ap);
        v16h a1 = load_afrag(ap + 16 * K);
        v16h b0 = load_bfrag(bq);
        v16h b1 = load_bfrag(bq + 16 * K);
        v16h b2 = load_bfrag(bq + 32 * K);
        v16h b3 = load_bfrag(bq + 48 * K);

        acc[0][0] = WMMA_F16(a0, b0, acc[0][0]);
        acc[1][0] = WMMA_F16(a1, b0, acc[1][0]);
        acc[0][1] = WMMA_F16(a0, b1, acc[0][1]);
        acc[1][1] = WMMA_F16(a1, b1, acc[1][1]);
        acc[0][2] = WMMA_F16(a0, b2, acc[0][2]);
        acc[1][2] = WMMA_F16(a1, b2, acc[1][2]);
        acc[0][3] = WMMA_F16(a0, b3, acc[0][3]);
        acc[1][3] = WMMA_F16(a1, b3, acc[1][3]);

        ap += 32;
        bq += 32;
    }

    // store: VGPR r holds row r (lanes 0-15) / row r+8 (lanes 16-31)
    const int rb = (lane & 16) ? 8 : 0;
#pragma unroll
    for (int half = 0; half < 2; ++half) {
        const int rbase = (mt << 5) + half * 16 + rb;
#pragma unroll
        for (int s = 0; s < 4; ++s) {
            const int c = col0 + s * 16;
#pragma unroll
            for (int r = 0; r < 8; ++r) {
                float v = acc[half][s][r];
                if (RELU) v = v > 0.f ? v : 0.f;
                size_t o = (size_t)(rbase + r) * NP + c;
                if (OUTF16) Ch[o] = (_Float16)v;
                else        Cf[o] = v;
            }
        }
    }
}

// ---------------- softmax over L*P=20 logits per (token, head) --------------
__global__ void softmax20_kernel(float* __restrict__ a)
{
    int i = blockIdx.x * blockDim.x + threadIdx.x;
    if (i >= M_ * H_) return;
    int tok = i >> 3, h = i & 7;
    float* p = a + (size_t)tok * NP_ATT + h * 20;
    float mx = -3.4e38f;
#pragma unroll
    for (int j = 0; j < 20; ++j) mx = fmaxf(mx, p[j]);
    float e[20]; float s = 0.f;
#pragma unroll
    for (int j = 0; j < 20; ++j) { e[j] = __expf(p[j] - mx); s += e[j]; }
    float inv = 1.f / s;
#pragma unroll
    for (int j = 0; j < 20; ++j) p[j] = e[j] * inv;
}

// ---------------- deformable bilinear sampling ------------------------------
// one wave per (token, head); lane = channel within head (HD=32)
__global__ void deform_sample_kernel(const float* __restrict__ offs,
                                     const float* __restrict__ attnw,
                                     const _Float16* __restrict__ value,
                                     const float* __restrict__ ref,
                                     _Float16* __restrict__ out)
{
    const int wv = (blockIdx.x * blockDim.x + threadIdx.x) >> 5;
    if (wv >= M_ * H_) return;
    const int lane = threadIdx.x & 31;
    const int tok  = wv >> 3;
    const int h    = wv & 7;
    const int bb   = tok / LQ_;

    const int dimH[L_]  = {100, 50, 25, 13, 7};
    const int start[L_] = {0, 10000, 12500, 13125, 13294};

    const float* __restrict__ orow = offs  + (size_t)tok * NP_OFF + h * (L_ * P_ * 2);
    const float* __restrict__ arow = attnw + (size_t)tok * NP_ATT + h * (L_ * P_);
    const float* __restrict__ rrow = ref   + (size_t)tok * (L_ * 2);

    float acc = 0.f;
#pragma unroll
    for (int l = 0; l < L_; ++l) {
        const int   Hh = dimH[l], Ww = dimH[l];
        const float fH = (float)Hh, fW = (float)Ww;
        const float rx = rrow[l * 2 + 0];
        const float ry = rrow[l * 2 + 1];
        const size_t vbase = ((size_t)bb * LQ_ + start[l]) * D_ + h * HD_ + lane;
#pragma unroll
        for (int p = 0; p < P_; ++p) {
            const float ox = orow[(l * P_ + p) * 2 + 0];
            const float oy = orow[(l * P_ + p) * 2 + 1];
            const float w  = arow[l * P_ + p];
            const float lx = rx + ox / fW;
            const float ly = ry + oy / fH;
            const float px = lx * fW - 0.5f;
            const float py = ly * fH - 0.5f;
            const float x0 = floorf(px), y0 = floorf(py);
            const float fx = px - x0,  fy = py - y0;
            const int x0i = (int)x0, y0i = (int)y0;
#pragma unroll
            for (int dy = 0; dy < 2; ++dy) {
#pragma unroll
                for (int dx = 0; dx < 2; ++dx) {
                    int xi = x0i + dx, yi = y0i + dy;
                    if (xi >= 0 && xi < Ww && yi >= 0 && yi < Hh) {
                        float bw = (dx ? fx : 1.f - fx) * (dy ? fy : 1.f - fy);
                        float v  = (float)value[vbase + (size_t)(yi * Ww + xi) * D_];
                        acc += w * bw * v;
                    }
                }
            }
        }
    }
    out[(size_t)tok * D_ + h * HD_ + lane] = (_Float16)acc;
}

// ---------------- wave32 sum reduction --------------------------------------
__device__ __forceinline__ float wave_sum(float v)
{
#pragma unroll
    for (int m = 16; m > 0; m >>= 1) v += __shfl_xor(v, m, 32);
    return v;
}

// ---------------- residual add + LayerNorm (one wave per token) -------------
template <bool F16OUT>
__global__ void add_ln_kernel(const float* __restrict__ a, const float* __restrict__ b,
                              const float* __restrict__ g, const float* __restrict__ be,
                              float* __restrict__ outf, _Float16* __restrict__ outh)
{
    const int wv = (blockIdx.x * blockDim.x + threadIdx.x) >> 5;
    if (wv >= M_) return;
    const int lane = threadIdx.x & 31;
    const size_t base = (size_t)wv * D_;

    float v[8];
    float s = 0.f;
#pragma unroll
    for (int j = 0; j < 8; ++j) {
        int c = j * 32 + lane;
        v[j] = a[base + c] + b[base + c];
        s += v[j];
    }
    s = wave_sum(s);
    const float mean = s * (1.f / D_);
    float q = 0.f;
#pragma unroll
    for (int j = 0; j < 8; ++j) { float d = v[j] - mean; q += d * d; }
    q = wave_sum(q);
    const float rstd = rsqrtf(q * (1.f / D_) + 1e-5f);
#pragma unroll
    for (int j = 0; j < 8; ++j) {
        int c = j * 32 + lane;
        float y = (v[j] - mean) * rstd * g[c] + be[c];
        outf[base + c] = y;
        if (F16OUT) outh[base + c] = (_Float16)y;
    }
}

// ---------------- host launcher ---------------------------------------------
extern "C" void kernel_launch(void* const* d_in, const int* in_sizes, int n_in,
                              void* d_out, int out_size, void* d_ws, size_t ws_size,
                              hipStream_t stream)
{
    const float* src   = (const float*)d_in[0];
    const float* pos   = (const float*)d_in[1];
    const float* ref   = (const float*)d_in[2];
    const float* W_off = (const float*)d_in[5];
    const float* b_off = (const float*)d_in[6];
    const float* W_att = (const float*)d_in[7];
    const float* b_att = (const float*)d_in[8];
    const float* W_val = (const float*)d_in[9];
    const float* b_val = (const float*)d_in[10];
    const float* W_out = (const float*)d_in[11];
    const float* b_out = (const float*)d_in[12];
    const float* g1    = (const float*)d_in[13];
    const float* be1   = (const float*)d_in[14];
    const float* W1    = (const float*)d_in[15];
    const float* bl1   = (const float*)d_in[16];
    const float* W2    = (const float*)d_in[17];
    const float* bl2   = (const float*)d_in[18];
    const float* g2    = (const float*)d_in[19];
    const float* be2   = (const float*)d_in[20];
    float* out = (float*)d_out;

    // ----- carve workspace -----
    char* p = (char*)d_ws;
    auto alloc = [&](size_t bytes) -> void* {
        void* r = (void*)p;
        p += (bytes + 255) & ~(size_t)255;
        return r;
    };
    _Float16* q16      = (_Float16*)alloc((size_t)MP_ * D_   * 2);
    _Float16* s16      = (_Float16*)alloc((size_t)MP_ * D_   * 2);
    _Float16* val16    = (_Float16*)alloc((size_t)MP_ * D_   * 2);
    _Float16* smp16    = (_Float16*)alloc((size_t)MP_ * D_   * 2);
    _Float16* x16      = (_Float16*)alloc((size_t)MP_ * D_   * 2);
    _Float16* h16      = (_Float16*)alloc((size_t)MP_ * DFF_ * 2);
    float*    offs     = (float*)   alloc((size_t)MP_ * NP_OFF * 4);
    float*    attnw    = (float*)   alloc((size_t)MP_ * NP_ATT * 4);
    float*    attn_out = (float*)   alloc((size_t)MP_ * D_   * 4);
    float*    xf32     = (float*)   alloc((size_t)MP_ * D_   * 4);
    float*    ffn      = (float*)   alloc((size_t)MP_ * D_   * 4);
    _Float16* Wt_off   = (_Float16*)alloc((size_t)NP_OFF * D_  * 2);
    _Float16* Wt_att   = (_Float16*)alloc((size_t)NP_ATT * D_  * 2);
    _Float16* Wt_val   = (_Float16*)alloc((size_t)D_ * D_      * 2);
    _Float16* Wt_out   = (_Float16*)alloc((size_t)D_ * D_      * 2);
    _Float16* Wt_1     = (_Float16*)alloc((size_t)DFF_ * D_    * 2);
    _Float16* Wt_2     = (_Float16*)alloc((size_t)D_ * DFF_    * 2);

    const int TB = 256;
    auto cdiv = [](int a, int b) { return (a + b - 1) / b; };

    // ----- weight convert/transpose -----
    convw_kernel<<<cdiv(NP_OFF * D_, TB), TB, 0, stream>>>(W_off, Wt_off, D_, NP_OFF, NP_OFF);
    convw_kernel<<<cdiv(NP_ATT * D_, TB), TB, 0, stream>>>(W_att, Wt_att, D_, NF_ATT, NP_ATT);
    convw_kernel<<<cdiv(D_ * D_, TB),     TB, 0, stream>>>(W_val, Wt_val, D_, D_, D_);
    convw_kernel<<<cdiv(D_ * D_, TB),     TB, 0, stream>>>(W_out, Wt_out, D_, D_, D_);
    convw_kernel<<<cdiv(DFF_ * D_, TB),   TB, 0, stream>>>(W1,    Wt_1,   D_,  DFF_, DFF_);
    convw_kernel<<<cdiv(D_ * DFF_, TB),   TB, 0, stream>>>(W2,    Wt_2,   DFF_, D_,  D_);

    // ----- prep activations -----
    prep_kernel<<<MP_, TB, 0, stream>>>(src, pos, q16, s16);

    const int MT = MP_ / 32;  // 1668 row tiles (32 rows per wave)

    // ----- value = src @ W_val + b_val  (f16 out, for sampling) -----
    {
        int waves = MT * (D_ / 64);
        gemm_wmma_kernel<D_, D_, false, true><<<cdiv(waves, 8), TB, 0, stream>>>(
            s16, Wt_val, b_val, nullptr, val16, D_, waves);
    }
    // ----- offs = q @ W_off + b_off  (f32 out) -----
    {
        int waves = MT * (NP_OFF / 64);
        gemm_wmma_kernel<D_, NP_OFF, false, false><<<cdiv(waves, 8), TB, 0, stream>>>(
            q16, Wt_off, b_off, offs, nullptr, NP_OFF, waves);
    }
    // ----- attn logits = q @ W_attn + b_attn  (f32 out) -----
    {
        int waves = MT * (NP_ATT / 64);
        gemm_wmma_kernel<D_, NP_ATT, false, false><<<cdiv(waves, 8), TB, 0, stream>>>(
            q16, Wt_att, b_att, attnw, nullptr, NF_ATT, waves);
    }
    // ----- softmax over 20 per (token, head) -----
    softmax20_kernel<<<cdiv(M_ * H_, TB), TB, 0, stream>>>(attnw);

    // ----- deformable sampling (one wave per token-head) -----
    deform_sample_kernel<<<cdiv(M_ * H_ * 32, TB), TB, 0, stream>>>(
        offs, attnw, val16, ref, smp16);

    // ----- attn_out = sampled @ W_out + b_out  (f32 out) -----
    {
        int waves = MT * (D_ / 64);
        gemm_wmma_kernel<D_, D_, false, false><<<cdiv(waves, 8), TB, 0, stream>>>(
            smp16, Wt_out, b_out, attn_out, nullptr, D_, waves);
    }
    // ----- x = LN(src + attn_out) -----
    add_ln_kernel<true><<<cdiv(M_ * 32, TB), TB, 0, stream>>>(
        src, attn_out, g1, be1, xf32, x16);

    // ----- h = relu(x @ W1 + bl1)  (f16 out) -----
    {
        int waves = MT * (DFF_ / 64);
        gemm_wmma_kernel<D_, DFF_, true, true><<<cdiv(waves, 8), TB, 0, stream>>>(
            x16, Wt_1, bl1, nullptr, h16, DFF_, waves);
    }
    // ----- ffn = h @ W2 + bl2  (f32 out) -----
    {
        int waves = MT * (D_ / 64);
        gemm_wmma_kernel<DFF_, D_, false, false><<<cdiv(waves, 8), TB, 0, stream>>>(
            h16, Wt_2, bl2, ffn, nullptr, D_, waves);
    }
    // ----- out = LN(x + ffn) -----
    add_ln_kernel<false><<<cdiv(M_ * 32, TB), TB, 0, stream>>>(
        xf32, ffn, g2, be2, out, nullptr);
}
